// FFLayer_14456859918441
// MI455X (gfx1250) — matrix operand order repaired and checked by reading, not compiled, not run on hardware
//
#include <hip/hip_runtime.h>
#include <math.h>

typedef __attribute__((ext_vector_type(16))) __bf16 v16bf;
typedef __attribute__((ext_vector_type(8)))  __bf16 v8bf;
typedef __attribute__((ext_vector_type(8)))  float  v8f;
typedef __attribute__((ext_vector_type(4)))  float  f32x4;

// ---------------- Row L2-norm kernel: one wave (32 lanes) per row ----------
__global__ __launch_bounds__(256) void row_norm_kernel(const float* __restrict__ X,
                                                       float* __restrict__ inv_norm,
                                                       int K) {
    const int row  = blockIdx.x * 8 + (threadIdx.x >> 5);
    const int lane = threadIdx.x & 31;
    const float* p = X + (size_t)row * K;
    float ss = 0.0f;
    for (int i = lane * 4; i < K; i += 32 * 4) {
        f32x4 v = *(const f32x4*)(p + i);
        ss += v[0] * v[0] + v[1] * v[1] + v[2] * v[2] + v[3] * v[3];
    }
    #pragma unroll
    for (int off = 16; off > 0; off >>= 1)
        ss += __shfl_xor(ss, off, 32);
    if (lane == 0)
        inv_norm[row] = 1.0f / (sqrtf(ss) + 1e-4f);
}

// ---------------- Fused GEMM + row-rescale + bias + ReLU -------------------
// out = relu( (x @ W^T) * inv_norm[row] + b )   [scale commutes with GEMM]
// Block: 512 threads (16 waves, 4x4). Tile: BM=128 x BN=256, BK=32.
// Each wave owns 32x64 = 2x4 WMMA 16x16x32 bf16 tiles, f32 accumulate.
#define BM  128
#define BN  256
#define BK  32
#define STR 40   // LDS row stride in bf16 elems (80B): pads banks, keeps 16B align

union B4 { __bf16 h[4]; unsigned long long u; };

__global__ __launch_bounds__(512) void ff_gemm_kernel(const float* __restrict__ X,
                                                      const float* __restrict__ W,
                                                      const float* __restrict__ bias,
                                                      const float* __restrict__ inv_norm,
                                                      float* __restrict__ out,
                                                      int K, int N) {
    __shared__ __attribute__((aligned(16))) __bf16 sA[2][BM * STR];  // 20 KB
    __shared__ __attribute__((aligned(16))) __bf16 sB[2][BN * STR];  // 40 KB

    const int tid  = threadIdx.x;
    const int m0   = blockIdx.y * BM;
    const int n0   = blockIdx.x * BN;

    const int lane = tid & 31;
    const int hl   = lane >> 4;    // lane-half: 0 or 1
    const int l15  = lane & 15;

    const int w    = tid >> 5;     // wave id 0..15
    const int wm   = w & 3;        // 4 waves across M
    const int wn   = w >> 2;       // 4 waves across N
    const int mb   = wm * 32;      // wave row base within tile
    const int nb   = wn * 64;      // wave col base within tile

    // Staging: chunks of f32x4. A: 128*32/4 = 1024 chunks (2/thread),
    //                           B: 256*32/4 = 2048 chunks (4/thread).
    // chunk c -> row = c>>3, kq = c&7 (which 4-float group of the 32 K).
    int a_row[2], a_kq[2], b_row[4], b_kq[4];
    #pragma unroll
    for (int i = 0; i < 2; ++i) { int c = tid + i * 512; a_row[i] = c >> 3; a_kq[i] = c & 7; }
    #pragma unroll
    for (int i = 0; i < 4; ++i) { int c = tid + i * 512; b_row[i] = c >> 3; b_kq[i] = c & 7; }

    v8f acc[2][4];
    #pragma unroll
    for (int tm = 0; tm < 2; ++tm)
        #pragma unroll
        for (int tn = 0; tn < 4; ++tn)
            #pragma unroll
            for (int j = 0; j < 8; ++j)
                acc[tm][tn][j] = 0.0f;

    auto stage_load = [&](int k0, f32x4 av[2], f32x4 wv[4]) {
        #pragma unroll
        for (int i = 0; i < 2; ++i)
            av[i] = *(const f32x4*)(X + (size_t)(m0 + a_row[i]) * K + k0 + a_kq[i] * 4);
        #pragma unroll
        for (int i = 0; i < 4; ++i)
            wv[i] = *(const f32x4*)(W + (size_t)(n0 + b_row[i]) * K + k0 + b_kq[i] * 4);
    };
    auto stage_store = [&](int buf, const f32x4 av[2], const f32x4 wv[4]) {
        #pragma unroll
        for (int i = 0; i < 2; ++i) {
            B4 t;
            #pragma unroll
            for (int e = 0; e < 4; ++e) t.h[e] = (__bf16)(av[i][e]);
            *reinterpret_cast<unsigned long long*>(&sA[buf][a_row[i] * STR + a_kq[i] * 4]) = t.u;
        }
        #pragma unroll
        for (int i = 0; i < 4; ++i) {
            B4 t;
            #pragma unroll
            for (int e = 0; e < 4; ++e) t.h[e] = (__bf16)(wv[i][e]);
            *reinterpret_cast<unsigned long long*>(&sB[buf][b_row[i] * STR + b_kq[i] * 4]) = t.u;
        }
    };

    // A fragment (16x32 bf16): lane-half 0 holds K 0-7 & 16-23, half 1 holds 8-15 & 24-31
    auto loadA = [&](const __bf16* s, int mrow) -> v16bf {
        const __bf16* p = s + (size_t)(mrow + l15) * STR + hl * 8;
        v8bf lo = *(const v8bf*)(p);        // K = hl*8 + 0..7
        v8bf hi = *(const v8bf*)(p + 16);   // K = 16 + hl*8 + 0..7
        return __builtin_shufflevector(lo, hi, 0,1,2,3,4,5,6,7,8,9,10,11,12,13,14,15);
    };
    // B fragment (32x16 bf16): lane = column n, lane-half 0 holds K 0-7 & 16-23, half 1 holds 8-15 & 24-31
    auto loadB = [&](const __bf16* s, int ncol) -> v16bf {
        const __bf16* p = s + (size_t)(ncol + l15) * STR + hl * 8;
        v8bf lo = *(const v8bf*)(p);        // K = hl*8 + 0..7
        v8bf hi = *(const v8bf*)(p + 16);   // K = 16 + hl*8 + 0..7
        return __builtin_shufflevector(lo, hi, 0,1,2,3,4,5,6,7,8,9,10,11,12,13,14,15);
    };

    // ---- software-pipelined K loop with double-buffered LDS ----
    f32x4 av[2], wv[4];
    stage_load(0, av, wv);
    stage_store(0, av, wv);

    const int nk = K / BK;
    for (int kb = 0; kb < nk; ++kb) {
        __syncthreads();                       // buf[kb&1] ready; other buf free
        const int cur = kb & 1;
        const bool pre = (kb + 1) < nk;
        if (pre) stage_load((kb + 1) * BK, av, wv);   // global loads hide behind WMMA

        v16bf afrag[2], bfrag[4];
        #pragma unroll
        for (int tm = 0; tm < 2; ++tm) afrag[tm] = loadA(sA[cur], mb + tm * 16);
        #pragma unroll
        for (int tn = 0; tn < 4; ++tn) bfrag[tn] = loadB(sB[cur], nb + tn * 16);

        #pragma unroll
        for (int tm = 0; tm < 2; ++tm)
            #pragma unroll
            for (int tn = 0; tn < 4; ++tn)
                acc[tm][tn] = __builtin_amdgcn_wmma_f32_16x16x32_bf16(
                    false, afrag[tm], false, bfrag[tn],
                    (short)0, acc[tm][tn], false, false);

        if (pre) stage_store(cur ^ 1, av, wv);
    }

    // ---- epilogue: per-row 1/||x|| rescale, bias, ReLU, store ----
    float rn[2][8];
    #pragma unroll
    for (int tm = 0; tm < 2; ++tm)
        #pragma unroll
        for (int j = 0; j < 8; ++j)
            rn[tm][j] = inv_norm[m0 + mb + tm * 16 + hl * 8 + j];

    #pragma unroll
    for (int tn = 0; tn < 4; ++tn) {
        const int col = n0 + nb + tn * 16 + l15;
        const float bb = bias[col];
        #pragma unroll
        for (int tm = 0; tm < 2; ++tm) {
            #pragma unroll
            for (int j = 0; j < 8; ++j) {
                const int row = m0 + mb + tm * 16 + hl * 8 + j;
                float v = acc[tm][tn][j] * rn[tm][j] + bb;
                out[(size_t)row * N + col] = v > 0.0f ? v : 0.0f;
            }
        }
    }
}

extern "C" void kernel_launch(void* const* d_in, const int* in_sizes, int n_in,
                              void* d_out, int out_size, void* d_ws, size_t ws_size,
                              hipStream_t stream) {
    const float* x = (const float*)d_in[0];
    const float* W = (const float*)d_in[1];
    const float* b = (const float*)d_in[2];
    float* out = (float*)d_out;

    const int OUT = in_sizes[2];               // 2048
    const int K   = in_sizes[1] / OUT;         // 2048
    const int B   = in_sizes[0] / K;           // 16384

    float* inv_norm = (float*)d_ws;            // B floats = 64 KB of scratch

    row_norm_kernel<<<B / 8, 256, 0, stream>>>(x, inv_norm, K);

    dim3 grid(OUT / BN, B / BM);               // (8, 128)
    ff_gemm_kernel<<<grid, 512, 0, stream>>>(x, W, b, inv_norm, out, K, OUT);
}
